// SelfAttention_35665408426211
// MI455X (gfx1250) — matrix operand order repaired and checked by reading, not compiled, not run on hardware
//
#include <hip/hip_runtime.h>
#include <hip/hip_fp16.h>
#include <math.h>

typedef _Float16 h8   __attribute__((ext_vector_type(8)));
typedef _Float16 v16h __attribute__((ext_vector_type(16)));
typedef float    v8f  __attribute__((ext_vector_type(8)));
typedef float    f4   __attribute__((ext_vector_type(4)));
typedef int      i4   __attribute__((vector_size(16)));

#define SEQ   8192
#define D_IN  256
#define D_OUT 64
#define KC    128               // keys staged in LDS per stage
#define NSTAGE (SEQ / KC)
#define L2E   1.44269504088896340736f

static __device__ inline v8f zero8() {
    v8f z;
#pragma unroll
    for (int i = 0; i < 8; ++i) z[i] = 0.0f;
    return z;
}

static __device__ inline v16h make16(h8 lo, h8 hi) {
    v16h r;
#pragma unroll
    for (int i = 0; i < 8; ++i) { r[i] = lo[i]; r[8 + i] = hi[i]; }
    return r;
}

// WMMA f32 = f16 x f16 + f32, 16x16x32
static __device__ inline v8f wmma16(v16h a, v16h b, v8f c) {
    return __builtin_amdgcn_wmma_f32_16x16x32_f16(false, a, false, b,
                                                  (short)0, c, false, false);
}

// ---- async global -> LDS 16-byte copy (ASYNCcnt path), with safe fallback ----
static __device__ inline void cp16(const _Float16* gsrc, _Float16* ldst) {
#if __has_builtin(__builtin_amdgcn_global_load_async_to_lds_b128)
    __builtin_amdgcn_global_load_async_to_lds_b128(
        (__attribute__((address_space(1))) i4*)gsrc,
        (__attribute__((address_space(3))) i4*)ldst, 0, 0);
#else
    *(h8*)ldst = *(const h8*)gsrc;
#endif
}

static __device__ inline void wait_async() {
#if __has_builtin(__builtin_amdgcn_s_wait_asynccnt)
    __builtin_amdgcn_s_wait_asynccnt(0);
#else
    asm volatile("s_wait_asynccnt 0" ::: "memory");
#endif
}

// A-layout 16-bit tile (16 x K=32) from row-major f16 (global or LDS).
// lane m (0..15): row m, K = {h*16 + 0..15}  (h = lane>>4)
static __device__ inline v16h loadA16(const _Float16* base, int stride, int lane) {
    int m = lane & 15, h = (lane >> 4) & 1;
    const _Float16* p = base + (size_t)m * stride + h * 16;
    h8 a = *(const h8*)p;
    h8 b = *(const h8*)(p + 8);
    return make16(a, b);
}

// B-layout 16-bit tile (K=32 x 16) whose column n is row n of a row-major f16 src.
// lane n: col n, K = {h*16 + 0..15}
static __device__ inline v16h loadB16(const _Float16* base, int stride, int lane) {
    int n = lane & 15, h = (lane >> 4) & 1;
    const _Float16* p = base + (size_t)n * stride + h * 16;
    h8 a = *(const h8*)p;
    h8 b = *(const h8*)(p + 8);
    return make16(a, b);
}

// A-layout tile of W^T (M=dims, K=features): element W[f][d0+m], W row-major [256][64] f32
static __device__ inline v16h loadWT_A(const float* W, int f0, int d0, int lane) {
    int m = lane & 15, h = (lane >> 4) & 1;
    const float* p = W + (size_t)(f0 + h * 16) * D_OUT + d0 + m;
    v16h r;
#pragma unroll
    for (int i = 0; i < 8; ++i) r[i] = (_Float16)p[i * D_OUT];
    p += 8 * D_OUT;
#pragma unroll
    for (int i = 0; i < 8; ++i) r[8 + i] = (_Float16)p[i * D_OUT];
    return r;
}

// B-layout tile of W (K=features, N=out dims): element W[f][n0+n]
static __device__ inline v16h loadW_B(const float* W, int f0, int n0, int lane) {
    int n = lane & 15, h = (lane >> 4) & 1;
    const float* p = W + (size_t)(f0 + h * 16) * D_OUT + n0 + n;
    v16h r;
#pragma unroll
    for (int i = 0; i < 16; ++i) r[i] = (_Float16)p[i * D_OUT];
    return r;
}

// x (f32 row-major [SEQ][256]) as B operand (x^T columns = x rows): lane n = row r0+n
static __device__ inline v16h loadX_B(const float* x, int r0, int f0, int lane) {
    int n = lane & 15, h = (lane >> 4) & 1;
    const float* p = x + (size_t)(r0 + n) * D_IN + f0 + h * 16;
    v16h r;
#pragma unroll
    for (int i = 0; i < 16; ++i) r[i] = (_Float16)p[i];
    return r;
}

// x as A operand: lane m = row r0+m, K = {f0+h*16+0..15}
static __device__ inline v16h loadX_A(const float* x, int r0, int f0, int lane) {
    int m = lane & 15, h = (lane >> 4) & 1;
    const float* p = x + (size_t)(r0 + m) * D_IN + f0 + h * 16;
    v16h r;
#pragma unroll
    for (int i = 0; i < 8; ++i) r[i] = (_Float16)p[i];
#pragma unroll
    for (int i = 0; i < 8; ++i) r[8 + i] = (_Float16)p[8 + i];
    return r;
}

// ---------------- Projection: q (scaled 1/8), k row-major f16; v stored transposed ----
__global__ __launch_bounds__(256) void proj_kernel(
    const float* __restrict__ x,
    const float* __restrict__ Wq, const float* __restrict__ Wk,
    const float* __restrict__ Wv,
    _Float16* __restrict__ qh, _Float16* __restrict__ kh,
    _Float16* __restrict__ vth) {
    const int lane = threadIdx.x & 31;
    const int wave = threadIdx.x >> 5;
    const int r0 = (blockIdx.x * 8 + wave) * 16;   // 16 rows of x per wave
    const int n = lane & 15, h = (lane >> 4) & 1;

    // ---- q^T / k^T orientation: D = W^T (dims x feat) x x^T (feat x queries) ----
    v16h xb[8];
#pragma unroll
    for (int kc = 0; kc < 8; ++kc) xb[kc] = loadX_B(x, r0, kc * 32, lane);

#pragma unroll
    for (int which = 0; which < 2; ++which) {
        const float* W = which ? Wk : Wq;
        _Float16* dst = which ? kh : qh;
        const float scale = which ? 1.0f : 0.125f;   // fold 1/sqrt(64) into q
#pragma unroll
        for (int dt = 0; dt < 4; ++dt) {
            v8f acc = zero8();
#pragma unroll
            for (int kc = 0; kc < 8; ++kc)
                acc = wmma16(loadWT_A(W, kc * 32, dt * 16, lane), xb[kc], acc);
            h8 o;
#pragma unroll
            for (int i = 0; i < 8; ++i) o[i] = (_Float16)(acc[i] * scale);
            *(h8*)(dst + (size_t)(r0 + n) * D_OUT + dt * 16 + h * 8) = o;
        }
    }

    // ---- v orientation: D = x (queries x feat) x W_v (feat x dims), store V^T ----
    v16h xa[8];
#pragma unroll
    for (int kc = 0; kc < 8; ++kc) xa[kc] = loadX_A(x, r0, kc * 32, lane);
#pragma unroll
    for (int nt = 0; nt < 4; ++nt) {
        v8f acc = zero8();
#pragma unroll
        for (int kc = 0; kc < 8; ++kc)
            acc = wmma16(xa[kc], loadW_B(Wv, kc * 32, nt * 16, lane), acc);
        h8 o;
#pragma unroll
        for (int i = 0; i < 8; ++i) o[i] = (_Float16)acc[i];
        *(h8*)(vth + (size_t)(nt * 16 + n) * SEQ + r0 + h * 8) = o;
    }
}

// Issue async copies of one 128-key stage: K rows (16KB) + V^T columns (16KB).
static __device__ inline void stage_issue(const _Float16* __restrict__ kh,
                                          const _Float16* __restrict__ vth,
                                          int k0, _Float16* bK, _Float16* bV,
                                          int tid) {
#pragma unroll
    for (int i = 0; i < 4; ++i) {        // K: KC*64 halves = 1024 chunks of 8 halves
        int c = tid + 256 * i;
        cp16(kh + (size_t)k0 * D_OUT + (size_t)c * 8, bK + (size_t)c * 8);
    }
#pragma unroll
    for (int i = 0; i < 4; ++i) {        // V^T: 64 rows x (KC/8)=16 chunks
        int c = tid + 256 * i;
        int d = c >> 4, j = c & 15;
        cp16(vth + (size_t)d * SEQ + k0 + j * 8, bV + (size_t)d * KC + j * 8);
    }
}

// ---------------- Flash attention: 8 waves share LDS-staged K/V chunks -------------
__global__ __launch_bounds__(256) void attn_kernel(
    const _Float16* __restrict__ qh, const _Float16* __restrict__ kh,
    const _Float16* __restrict__ vth, float* __restrict__ out) {
    __shared__ _Float16 sK[2][KC * D_OUT];   // [key][dim], 16KB per buffer
    __shared__ _Float16 sV[2][D_OUT * KC];   // [dim][key], 16KB per buffer

    const int tid = threadIdx.x;
    const int lane = tid & 31;
    const int wave = tid >> 5;
    const int q0 = (blockIdx.x * 8 + wave) * 16;
    const int n = lane & 15, h = (lane >> 4) & 1;

    // Q as B operand (features 0-31 and 32-63), loaded once
    const v16h qb0 = loadB16(qh + (size_t)q0 * D_OUT + 0, D_OUT, lane);
    const v16h qb1 = loadB16(qh + (size_t)q0 * D_OUT + 32, D_OUT, lane);

    v8f acc[4];
#pragma unroll
    for (int dt = 0; dt < 4; ++dt) acc[dt] = zero8();
    float mrun = -INFINITY, lrun = 0.0f;

    stage_issue(kh, vth, 0, sK[0], sV[0], tid);

    for (int s = 0; s < NSTAGE; ++s) {
        wait_async();          // this thread's stage-s copies have landed in LDS
        __syncthreads();       // ... and everyone else's too
        if (s + 1 < NSTAGE)
            stage_issue(kh, vth, (s + 1) * KC, sK[(s + 1) & 1], sV[(s + 1) & 1], tid);

        const _Float16* bK = sK[s & 1];
        const _Float16* bV = sV[s & 1];

#pragma unroll
        for (int sub = 0; sub < KC / 32; ++sub) {
            const _Float16* kb = bK + (size_t)(sub * 32) * D_OUT;

            // S^T = K x Q^T : tile0 = keys +0..15, tile1 = keys +16..31
            v8f s0 = zero8(), s1 = zero8();
            s0 = wmma16(loadA16(kb, D_OUT, lane), qb0, s0);
            s0 = wmma16(loadA16(kb + 32, D_OUT, lane), qb1, s0);
            s1 = wmma16(loadA16(kb + 16 * D_OUT, D_OUT, lane), qb0, s1);
            s1 = wmma16(loadA16(kb + 16 * D_OUT + 32, D_OUT, lane), qb1, s1);

            // ---- online softmax for query q0+n (16 scores here, 16 in lane^16) ----
            float cm = -INFINITY;
#pragma unroll
            for (int r = 0; r < 8; ++r) cm = fmaxf(cm, fmaxf(s0[r], s1[r]));
            cm = fmaxf(cm, __shfl_xor(cm, 16, 32));
            const float mn = fmaxf(mrun, cm);
            const float corr = exp2f((mrun - mn) * L2E);

            float p0[8], p1[8], ls = 0.0f;
#pragma unroll
            for (int r = 0; r < 8; ++r) {
                p0[r] = exp2f((s0[r] - mn) * L2E);
                p1[r] = exp2f((s1[r] - mn) * L2E);
                ls += p0[r] + p1[r];
            }
            ls += __shfl_xor(ls, 16, 32);
            lrun = lrun * corr + ls;
            mrun = mn;
#pragma unroll
            for (int dt = 0; dt < 4; ++dt)
#pragma unroll
                for (int r = 0; r < 8; ++r) acc[dt][r] *= corr;

            // ---- pack P into B-operand layout (lane = query column) ----
            // lo lane holds keys {0..7}(p0) & {16..23}(p1); hi lane {8..15} & {24..31}
            unsigned d0[4], d1[4];
#pragma unroll
            for (int j = 0; j < 4; ++j) {
                union { _Float16 hh[2]; unsigned u; } t;
                t.hh[0] = (_Float16)p0[2 * j]; t.hh[1] = (_Float16)p0[2 * j + 1];
                d0[j] = t.u;
                t.hh[0] = (_Float16)p1[2 * j]; t.hh[1] = (_Float16)p1[2 * j + 1];
                d1[j] = t.u;
            }
            unsigned x0[4], x1[4];
#pragma unroll
            for (int j = 0; j < 4; ++j) {
                x0[j] = __shfl_xor(d0[j], 16, 32);
                x1[j] = __shfl_xor(d1[j], 16, 32);
            }
            union { unsigned u[8]; v16h v; } pb;
#pragma unroll
            for (int j = 0; j < 4; ++j) {
                pb.u[j]     = h ? x1[j] : d0[j];  // lo: keys 0..7  | hi: keys 16..23
                pb.u[4 + j] = h ? d1[j] : x0[j];  // lo: keys 8..15 | hi: keys 24..31
            }

            // ---- O^T += V^T x P^T : 4 dim-tiles ----
#pragma unroll
            for (int dt = 0; dt < 4; ++dt) {
                v16h va = loadA16(bV + (size_t)(dt * 16) * KC + sub * 32, KC, lane);
                acc[dt] = wmma16(va, pb.v, acc[dt]);
            }
        }
        __syncthreads();       // all waves done reading buffer s&1
    }

    const float inv = 1.0f / lrun;
#pragma unroll
    for (int dt = 0; dt < 4; ++dt) {
        f4 o0, o1;
#pragma unroll
        for (int i = 0; i < 4; ++i) { o0[i] = acc[dt][i] * inv; o1[i] = acc[dt][4 + i] * inv; }
        float* p = out + (size_t)(q0 + n) * D_OUT + dt * 16 + h * 8;
        *(f4*)p = o0;
        *(f4*)(p + 4) = o1;
    }
}

extern "C" void kernel_launch(void* const* d_in, const int* in_sizes, int n_in,
                              void* d_out, int out_size, void* d_ws, size_t ws_size,
                              hipStream_t stream) {
    const float* x  = (const float*)d_in[0];
    const float* Wq = (const float*)d_in[1];
    const float* Wk = (const float*)d_in[2];
    const float* Wv = (const float*)d_in[3];

    _Float16* qh  = (_Float16*)d_ws;            // [SEQ][64] f16, pre-scaled by 1/8
    _Float16* kh  = qh + (size_t)SEQ * D_OUT;   // [SEQ][64] f16
    _Float16* vth = kh + (size_t)SEQ * D_OUT;   // [64][SEQ] f16 (V transposed)

    proj_kernel<<<SEQ / 16 / 8, 256, 0, stream>>>(x, Wq, Wk, Wv, qh, kh, vth);
    attn_kernel<<<SEQ / 16 / 8, 256, 0, stream>>>(qh, kh, vth, (float*)d_out);
}
